// TopKExpertsSparseLinear_40785009442943
// MI455X (gfx1250) — matrix operand order repaired and checked by reading, not compiled, not run on hardware
//
#include <hip/hip_runtime.h>
#include <stdint.h>

// Problem constants (match reference)
#define TT   16384            // B*S*K tokens
#define EE   8
#define DIN  1024
#define DOUT 1024

typedef __attribute__((ext_vector_type(16))) __bf16 v16bf;
typedef __attribute__((ext_vector_type(8)))  float  v8f;
typedef __attribute__((ext_vector_type(4)))  unsigned int v4u;

union Frag { v16bf v; v4u u[2]; };

// Workspace layout (~16.6 MB):
//   [0,128)                  : int cnt[EE] (padded)
//   [128, 128+EE*TT*4)       : int list[EE][TT] (token ids per expert)
//   [WS_WT_OFF, +16MB)       : bf16 Wf in B-fragment order:
//                              [e][s2=k/32][n0=o/16][lane(32)][j(16)]
#define WS_CNT_OFF  0
#define WS_LIST_OFF 128
#define WS_WT_OFF   (128 + EE * TT * 4)

// ---------------- pass 1: bin tokens by expert ----------------
__global__ void moe_bin_kernel(const int* __restrict__ idx,
                               int* __restrict__ cnt,
                               int* __restrict__ list) {
  int t = blockIdx.x * blockDim.x + threadIdx.x;
  if (t < TT) {
    int e = idx[t] & (EE - 1);
    int p = atomicAdd(&cnt[e], 1);
    list[e * TT + p] = t;
  }
}

// ------- pass 2: repack W[o][e][d] fp32 -> Wf (bf16, B-fragment order) -------
// Fragment layout (this code's contiguous k order, 16-bit B 32x16): lane l -> column n = l&15,
// K base = (l>>4)*16; element j -> k = kbase + j.
__global__ void moe_wconv_kernel(const float* __restrict__ W,
                                 __bf16* __restrict__ Wf) {
  size_t i = (size_t)blockIdx.x * blockDim.x + threadIdx.x;  // 8*32*64*32*16
  int j  = (int)(i & 15);
  int l  = (int)((i >> 4) & 31);
  int n0 = (int)((i >> 9) & 63);
  int s2 = (int)((i >> 15) & 31);
  int e  = (int)(i >> 20);
  int o  = n0 * 16 + (l & 15);
  int d  = s2 * 32 + ((l >> 4) << 4) + j;
  Wf[i] = (__bf16)W[(size_t)o * (EE * DIN) + (size_t)e * DIN + d];
}

// ---------------- pass 3: per-expert tiled GEMM via WMMA bf16 ----------------
// block = 256 threads = 8 waves; tile = 32 tokens x 128 outputs; K-step = 64.
// Manually software-pipelined 2 deep: ping-pong LDS A buffers (compile-time
// constant) + ping-pong B-fragment register sets -> no accumulator copies.
__global__ __launch_bounds__(256)
void moe_gemm_kernel(const float*  __restrict__ x,
                     const __bf16* __restrict__ Wf,
                     const float*  __restrict__ bias,
                     const int*    __restrict__ cnt,
                     const int*    __restrict__ list,
                     float*        __restrict__ out) {
  const int e     = blockIdx.z;
  const int chunk = blockIdx.y;
  const int nBase = blockIdx.x * 128;
  const int count = cnt[e];
  if (chunk * 32 >= count) return;   // uniform early-exit

  __shared__ int sTokLd[32];                     // clamped (always valid)
  __shared__ int sTokSt[32];                     // -1 => padded row, no store
  __shared__ __align__(16) __bf16 As[2][32][72]; // double buffer, padded rows

  const int tid = threadIdx.x;
  if (tid < 32) {
    int r     = chunk * 32 + tid;
    int first = list[e * TT + chunk * 32];       // exists: chunk*32 < count
    int tv    = (r < count) ? list[e * TT + r] : -1;
    sTokSt[tid] = tv;
    sTokLd[tid] = (tv >= 0) ? tv : first;        // branch-free staging loads
  }
  __syncthreads();

  const int wave    = tid >> 5;
  const int lane    = tid & 31;
  const int halfSel = lane >> 4;
  const int ln      = lane & 15;
  const int n0      = (nBase >> 4) + wave;       // this wave's 16-col tile

  // v4u index into Wf: ((e*32 + s2)*64 + n0)*32*2 + lane*2
  const v4u* bW = (const v4u*)Wf +
                  (size_t)e * 131072 + ((size_t)n0 * 32 + (size_t)lane) * 2;

  // A staging coords: 8 bf16 per thread, per-thread source row pointer
  const int arow = tid >> 3;
  const int acol = (tid & 7) * 8;
  const float* xp = x + (size_t)sTokLd[arow] * DIN + acol;

  v8f c0 = {}, c1 = {};
  float4 xa, xb;

  auto loadX = [&](int s) {
    const float4* xq = (const float4*)(xp + s * 64);
    xa = xq[0];
    xb = xq[1];
  };

  auto loadB = [&](int s, Frag& f0, Frag& f1) {
    const v4u* bp = bW + (size_t)(2 * s) * 4096;
    f0.u[0] = bp[0];    f0.u[1] = bp[1];
    f1.u[0] = bp[4096]; f1.u[1] = bp[4097];
  };

  auto stageA = [&](int buf) {
    union { __bf16 h[8]; v4u u; } pk;
    pk.h[0] = (__bf16)xa.x; pk.h[1] = (__bf16)xa.y;
    pk.h[2] = (__bf16)xa.z; pk.h[3] = (__bf16)xa.w;
    pk.h[4] = (__bf16)xb.x; pk.h[5] = (__bf16)xb.y;
    pk.h[6] = (__bf16)xb.z; pk.h[7] = (__bf16)xb.w;
    *(v4u*)&As[buf][arow][acol] = pk.u;
  };

  auto computeTile = [&](int buf, const Frag& f0, const Frag& f1) {
    const int kA = halfSel * 16;  // A layout: k=kA+j
    const __bf16* b0 = &As[buf][ln][0];
    const __bf16* b1 = &As[buf][16 + ln][0];
    Frag fa00, fa01, fa10, fa11;
    fa00.u[0] = *(const v4u*)(b0 + kA);      fa00.u[1] = *(const v4u*)(b0 + kA + 8);
    fa10.u[0] = *(const v4u*)(b1 + kA);      fa10.u[1] = *(const v4u*)(b1 + kA + 8);
    fa01.u[0] = *(const v4u*)(b0 + 32 + kA); fa01.u[1] = *(const v4u*)(b0 + 32 + kA + 8);
    fa11.u[0] = *(const v4u*)(b1 + 32 + kA); fa11.u[1] = *(const v4u*)(b1 + 32 + kA + 8);
    c0 = __builtin_amdgcn_wmma_f32_16x16x32_bf16(false, fa00.v, false, f0.v,
                                                 (short)0, c0, false, false);
    c1 = __builtin_amdgcn_wmma_f32_16x16x32_bf16(false, fa10.v, false, f0.v,
                                                 (short)0, c1, false, false);
    c0 = __builtin_amdgcn_wmma_f32_16x16x32_bf16(false, fa01.v, false, f1.v,
                                                 (short)0, c0, false, false);
    c1 = __builtin_amdgcn_wmma_f32_16x16x32_bf16(false, fa11.v, false, f1.v,
                                                 (short)0, c1, false, false);
  };

  Frag fbA0, fbA1, fbB0, fbB1;

  // ---- prologue: load + stage step 0 ----
  loadX(0);
  loadB(0, fbA0, fbA1);
  stageA(0);
  __syncthreads();

  // ---- main loop: steps 0..13 in pairs, pipelined one step ahead ----
#pragma unroll 1
  for (int s = 0; s < (DIN / 64) - 2; s += 2) {
    // even step s: compute buf0 with set A, stage s+1 -> buf1, refill set B
    loadX(s + 1);
    loadB(s + 1, fbB0, fbB1);
    computeTile(0, fbA0, fbA1);
    stageA(1);
    __syncthreads();

    // odd step s+1: compute buf1 with set B, stage s+2 -> buf0, refill set A
    loadX(s + 2);
    loadB(s + 2, fbA0, fbA1);
    computeTile(1, fbB0, fbB1);
    stageA(0);
    __syncthreads();
  }

  // ---- epilogue steps 14 and 15 (no further prefetch) ----
  loadX(15);
  loadB(15, fbB0, fbB1);
  computeTile(0, fbA0, fbA1);
  stageA(1);
  __syncthreads();
  computeTile(1, fbB0, fbB1);

  // ---- bias add + scatter to original token rows ----
  const int   o   = nBase + wave * 16 + ln;
  const float bv  = bias[o * EE + e];
  const int   mhi = halfSel << 3;   // C/D: VGPR v -> M = v (+8 for upper lanes)
#pragma unroll
  for (int v = 0; v < 8; ++v) {
    int m  = v + mhi;
    int t0 = sTokSt[m];
    if (t0 >= 0) out[(size_t)t0 * DOUT + o] = c0[v] + bv;
    int t1 = sTokSt[16 + m];
    if (t1 >= 0) out[(size_t)t1 * DOUT + o] = c1[v] + bv;
  }
}

// ---------------- launcher ----------------
extern "C" void kernel_launch(void* const* d_in, const int* in_sizes, int n_in,
                              void* d_out, int out_size, void* d_ws, size_t ws_size,
                              hipStream_t stream) {
  (void)in_sizes; (void)n_in; (void)out_size; (void)ws_size;
  const float* x    = (const float*)d_in[0];
  const int*   idx  = (const int*)d_in[1];
  const float* W    = (const float*)d_in[2];
  const float* bias = (const float*)d_in[3];
  float*       out  = (float*)d_out;

  char*   ws   = (char*)d_ws;
  int*    cnt  = (int*)(ws + WS_CNT_OFF);
  int*    list = (int*)(ws + WS_LIST_OFF);
  __bf16* Wf   = (__bf16*)(ws + WS_WT_OFF);

  hipMemsetAsync(cnt, 0, 128, stream);
  moe_bin_kernel<<<(TT + 255) / 256, 256, 0, stream>>>(idx, cnt, list);
  moe_wconv_kernel<<<(EE * DIN * DOUT) / 256, 256, 0, stream>>>(W, Wf);
  dim3 grid(DOUT / 128, TT / 32, EE);
  moe_gemm_kernel<<<grid, 256, 0, stream>>>(x, Wf, bias, cnt, list, out);
}
